// Decoder_2_48593259987149
// MI455X (gfx1250) — matrix operand order repaired and checked by reading, not compiled, not run on hardware
//
#include <hip/hip_runtime.h>
#include <hip/hip_bf16.h>

typedef __bf16 bf16_t;
typedef bf16_t v16bf __attribute__((ext_vector_type(16)));
typedef float  v8f   __attribute__((ext_vector_type(8)));

#define B_   64
#define T_   256
#define DIN  600
#define DINP 608
#define H_   1600
#define G4H  6400
#define D1_  600
#define D1P  608
#define D2_  3
#define BT   (B_*T_)   // 16384

union Frag { v16bf v; uint4 q[2]; };

__device__ inline v8f zero8(){ v8f z;
#pragma unroll
  for (int i = 0; i < 8; ++i) z[i] = 0.f;
  return z; }

__device__ inline bf16_t f2bf(float x){ return (bf16_t)x; }

__device__ inline float sigm(float x){ return 1.f/(1.f + __expf(-x)); }
__device__ inline float tanhfast(float x){
  float t = __expf(-2.f*fabsf(x));
  float r = (1.f - t)/(1.f + t);
  return x < 0.f ? -r : r;
}

// A fragment (16x32 bf16, row-major source). lane = m%16 + 16*half.
// element j: k = k0 + 16*half + j
__device__ inline v16bf load_a_rm(const bf16_t* A, long lda, int k0, int lane){
  int m = lane & 15, h = lane >> 4;
  const bf16_t* p = A + (long)m*lda + k0 + 16*h;
  Frag f;
  f.q[0] = *reinterpret_cast<const uint4*>(p);
  f.q[1] = *reinterpret_cast<const uint4*>(p + 8);
  return f.v;
}

// B fragment from packed tiles: tile = 512 bf16, lane-major contiguous.
__device__ inline v16bf load_b_pk(const bf16_t* Bp, long tile, int lane){
  const bf16_t* p = Bp + (tile*32 + (long)lane)*16;
  Frag f;
  f.q[0] = *reinterpret_cast<const uint4*>(p);
  f.q[1] = *reinterpret_cast<const uint4*>(p + 8);
  return f.v;
}

// B fragment where B = M^T and M is row-major [N x K]: B[k][n] = M[n0+n][k0+k].
// lane = n + 16*half; element j -> M[n0+n][k0 + 16*half + j]  (contiguous!)
__device__ inline v16bf load_b_rmT(const bf16_t* M, long ld, int n0, int k0, int lane){
  int n = lane & 15, h = lane >> 4;
  const bf16_t* p = M + (long)(n0 + n)*ld + k0 + 16*h;
  Frag f;
  f.q[0] = *reinterpret_cast<const uint4*>(p);
  f.q[1] = *reinterpret_cast<const uint4*>(p + 8);
  return f.v;
}

#define WMMA(a,b,c) __builtin_amdgcn_wmma_f32_16x16x32_bf16(false,(a),false,(b),(short)0,(c),false,false)

// ---------------- pack / convert kernels ----------------

__global__ void cvt_kernel(const float* in, bf16_t* out, long n){
  long i = (long)blockIdx.x*blockDim.x + threadIdx.x;
  if (i < n) out[i] = f2bf(in[i]);
}

__global__ void copyf_kernel(const float* in, float* out, long n){
  long i = (long)blockIdx.x*blockDim.x + threadIdx.x;
  if (i < n) out[i] = in[i];
}

__global__ void cvt_pad_rows_kernel(const float* in, bf16_t* out, long rows, int K, int Kpad){
  long i = (long)blockIdx.x*blockDim.x + threadIdx.x;
  if (i >= rows*(long)Kpad) return;
  long r = i / Kpad; int k = (int)(i % Kpad);
  out[i] = (k < K) ? f2bf(in[r*(long)K + k]) : (bf16_t)0.f;
}

// Pack row-major W[K x N] (f32) into WMMA B-fragment tile order (Kpad x Npad).
__global__ void pack_b_kernel(const float* W, bf16_t* out, int K, int N, int Kpad, int Npad){
  long i = (long)blockIdx.x*blockDim.x + threadIdx.x;
  long total = (long)(Kpad/32)*(Npad/16)*512;
  if (i >= total) return;
  int j = (int)(i & 15);
  int lane = (int)((i >> 4) & 31);
  long tile = i >> 9;
  int NT = Npad/16;
  int kt = (int)(tile / NT), nt = (int)(tile % NT);
  int k = kt*32 + 16*(lane >> 4) + j;
  int n = nt*16 + (lane & 15);
  float v = (k < K && n < N) ? W[(long)k*N + n] : 0.f;
  out[i] = f2bf(v);
}

// Pack H_value[b][256][1600] per batch into B-fragment tile order (K=256, N=1600).
__global__ void pack_hv_kernel(const float* Hv, bf16_t* out){
  long i = (long)blockIdx.x*blockDim.x + threadIdx.x;
  long total = (long)B_*8*100*512;
  if (i >= total) return;
  int j = (int)(i & 15);
  int lane = (int)((i >> 4) & 31);
  long tile = i >> 9;           // b*800 + kt*100 + nt
  int b  = (int)(tile / 800);
  int rem = (int)(tile % 800);
  int kt = rem / 100, nt = rem % 100;
  int k = kt*32 + 16*(lane >> 4) + j;   // key index (<256)
  int n = nt*16 + (lane & 15);          // hidden index (<1600)
  out[i] = f2bf(Hv[(((long)b*T_) + k)*H_ + n]);
}

// ---------------- xproj = y @ Wx + b ----------------
// grid (BT/64, G4H/64), block 128 (4 waves, each wave: 16 rows x 64 cols)
__global__ __launch_bounds__(128) void gemm_xproj_kernel(
    const bf16_t* __restrict__ yb, const bf16_t* __restrict__ WxP,
    const float* __restrict__ bias, float* __restrict__ xproj){
  int wave = threadIdx.x >> 5, lane = threadIdx.x & 31;
  int m0 = blockIdx.x*64 + wave*16;
  int nblk = blockIdx.y;
  const int NT = G4H/16, KT = DINP/32;
  const bf16_t* Arow = yb + (long)m0*DINP;
  v8f acc[4];
#pragma unroll
  for (int i = 0; i < 4; ++i) acc[i] = zero8();
  for (int kt = 0; kt < KT; ++kt){
    v16bf a = load_a_rm(Arow, DINP, kt*32, lane);
#pragma unroll
    for (int i = 0; i < 4; ++i){
      v16bf bb = load_b_pk(WxP, (long)kt*NT + nblk*4 + i, lane);
      acc[i] = WMMA(a, bb, acc[i]);
    }
  }
  int mrow = m0 + 8*(lane >> 4);
  int ncol = nblk*64 + (lane & 15);
#pragma unroll
  for (int i = 0; i < 4; ++i){
    int n = ncol + i*16;
    float bv = bias[n];
#pragma unroll
    for (int r = 0; r < 8; ++r)
      xproj[(long)(mrow + r)*G4H + n] = acc[i][r] + bv;
  }
}

// ---------------- one LSTM time step ----------------
// grid (H_/16 = 100), block 512 (16 waves). Block owns 16 hidden units x all 4
// gates x all 64 batches. Wave w: gate (w&3), batch tile (w>>2): ONE accumulator,
// 50-step K chain -> maximal WMMA pipelining; z exchanged via LDS for gate fusion.
__global__ __launch_bounds__(512) void lstm_step_kernel(
    const bf16_t* __restrict__ hprev, long lda,
    const bf16_t* __restrict__ WhP, const float* __restrict__ xproj,
    float* __restrict__ cbuf, bf16_t* __restrict__ Sbf,
    float* __restrict__ hfin, int t){
  __shared__ __align__(16) float zs[4][4][16][16];   // [mtile][gate][m][n] = 16 KB
  int tid  = threadIdx.x;
  int wave = tid >> 5, lane = tid & 31;
  int g  = wave & 3;        // gate
  int mt = wave >> 2;       // batch tile (0..3)
  int m0 = mt*16;
  int u0 = blockIdx.x*16;
  const int NT = G4H/16;
  const bf16_t* Arow = hprev + (long)m0*lda;
  const long btile = (long)g*(H_/16) + blockIdx.x;
  v8f acc = zero8();
  for (int kt = 0; kt < H_/32; ++kt){
    // prefetch next K-tile of packed Wh (dominant L2 stream in the recurrence)
    int ktn = (kt < H_/32 - 1) ? kt + 1 : kt;
    __builtin_prefetch(WhP + (((long)ktn*NT + btile)*32 + lane)*16, 0, 1);
    v16bf a  = load_a_rm(Arow, lda, kt*32, lane);
    v16bf bb = load_b_pk(WhP, (long)kt*NT + btile, lane);
    acc = WMMA(a, bb, acc);
  }
  {
    int half = lane >> 4, n = lane & 15;
#pragma unroll
    for (int r = 0; r < 8; ++r)
      zs[mt][g][r + 8*half][n] = acc[r];
  }
  __syncthreads();
  // gate fusion: 64 batches x 16 units = 1024 outputs, 512 threads -> 2 each
  for (int e = tid; e < B_*16; e += 512){
    int m  = e >> 4;          // batch
    int ul = e & 15;          // local unit
    int u  = u0 + ul;
    long xbase = ((long)m*T_ + t)*G4H;
    float zi = zs[m >> 4][0][m & 15][ul] + xproj[xbase + u];
    float zf = zs[m >> 4][1][m & 15][ul] + xproj[xbase + H_   + u];
    float zg = zs[m >> 4][2][m & 15][ul] + xproj[xbase + 2*H_ + u];
    float zo = zs[m >> 4][3][m & 15][ul] + xproj[xbase + 3*H_ + u];
    float ii = sigm(zi), ff = sigm(zf), gg = tanhfast(zg), oo = sigm(zo);
    long cu = (long)m*H_ + u;
    float c = ff*cbuf[cu] + ii*gg;
    cbuf[cu] = c;
    float h = oo*tanhfast(c);
    Sbf[((long)m*T_ + t)*H_ + u] = f2bf(h);
    hfin[cu] = h;
  }
}

// ---------------- attention: per (batch, 16-query tile) ----------------
// grid (B_, T_/16), block 128
__global__ __launch_bounds__(128) void attn_kernel(
    const bf16_t* __restrict__ s0b, const bf16_t* __restrict__ Sbf,
    const bf16_t* __restrict__ Hvb, const bf16_t* __restrict__ HvP,
    bf16_t* __restrict__ Abf){
  __shared__ __align__(16) float  sc[16][256];
  __shared__ __align__(16) bf16_t P[16][256];
  int b = blockIdx.x, qb = blockIdx.y;
  int wave = threadIdx.x >> 5, lane = threadIdx.x & 31;
  int mloc = lane & 15, half = lane >> 4;
  int tq = qb*16 + mloc;
  // shifted decoder state: S_key[t] = (t==0) ? s0 : S[t-1]
  const bf16_t* qrow = (tq == 0) ? (s0b + (long)b*H_)
                                 : (Sbf + ((long)b*T_ + tq - 1)*H_);
  const bf16_t* HvbB = Hvb + (long)b*T_*H_;
  // Phase 1: scores[16][256] = Q @ Hv^T  (K = 1600)
  v8f acc[4];
#pragma unroll
  for (int i = 0; i < 4; ++i) acc[i] = zero8();
  for (int kt = 0; kt < H_/32; ++kt){
    Frag fa;
    const bf16_t* pa = qrow + kt*32 + 16*half;
    fa.q[0] = *reinterpret_cast<const uint4*>(pa);
    fa.q[1] = *reinterpret_cast<const uint4*>(pa + 8);
#pragma unroll
    for (int i = 0; i < 4; ++i){
      v16bf bb = load_b_rmT(HvbB, H_, wave*64 + i*16, kt*32, lane);
      acc[i] = WMMA(fa.v, bb, acc[i]);
    }
  }
#pragma unroll
  for (int i = 0; i < 4; ++i){
    int n = wave*64 + i*16 + (lane & 15);
#pragma unroll
    for (int r = 0; r < 8; ++r)
      sc[r + 8*half][n] = acc[i][r];
  }
  __syncthreads();
  // softmax over keys, one row per thread (16 rows)
  if (threadIdx.x < 16){
    int rq = threadIdx.x;
    float mx = -3.4e38f;
    for (int k = 0; k < 256; ++k) mx = fmaxf(mx, sc[rq][k]);
    float s = 0.f;
    for (int k = 0; k < 256; ++k){ float e = __expf(sc[rq][k]-mx); sc[rq][k] = e; s += e; }
    float inv = 1.f/s;
    for (int k = 0; k < 256; ++k) P[rq][k] = f2bf(sc[rq][k]*inv);
  }
  __syncthreads();
  // Phase 2: A = P @ Hv  (K = 256, N = 1600); each wave: 25 n-tiles
  for (int nti = 0; nti < 25; ++nti){
    int nt = wave*25 + nti;
    v8f a2 = zero8();
#pragma unroll
    for (int kt = 0; kt < 8; ++kt){
      Frag fa;
      const bf16_t* pp = &P[lane & 15][kt*32 + 16*half];
      fa.q[0] = *reinterpret_cast<const uint4*>(pp);
      fa.q[1] = *reinterpret_cast<const uint4*>(pp + 8);
      v16bf bb = load_b_pk(HvP + (long)b*(8*100*512), kt*100 + nt, lane);
      a2 = WMMA(fa.v, bb, a2);
    }
    int n = nt*16 + (lane & 15);
#pragma unroll
    for (int r = 0; r < 8; ++r){
      int tl = r + 8*half;
      Abf[(((long)b*T_ + qb*16 + tl))*H_ + n] = f2bf(a2[r]);
    }
  }
}

// ---------------- x1 = concat(S, A) @ W1 + b1 ----------------
// grid (BT/64, D1P/16), block 128. wave w: rows 16w..; one n-tile per block.
__global__ __launch_bounds__(128) void gemm_w1_kernel(
    const bf16_t* __restrict__ Sbf, const bf16_t* __restrict__ Abf,
    const bf16_t* __restrict__ W1P, const float* __restrict__ b1,
    float* __restrict__ x1){
  int wave = threadIdx.x >> 5, lane = threadIdx.x & 31;
  int m0 = blockIdx.x*64 + wave*16;
  int nt = blockIdx.y;
  const int NT = D1P/16;
  v8f acc = zero8();
  for (int kt = 0; kt < (2*H_)/32; ++kt){
    int k0 = kt*32;
    v16bf a = (k0 < H_) ? load_a_rm(Sbf + (long)m0*H_, H_, k0, lane)
                        : load_a_rm(Abf + (long)m0*H_, H_, k0 - H_, lane);
    v16bf bb = load_b_pk(W1P, (long)kt*NT + nt, lane);
    acc = WMMA(a, bb, acc);
  }
  int n = nt*16 + (lane & 15);
  if (n < D1_){
    float bv = b1[n];
    int mrow = m0 + 8*(lane >> 4);
#pragma unroll
    for (int r = 0; r < 8; ++r)
      x1[(long)(mrow + r)*D1_ + n] = acc[r] + bv;
  }
}

// ---------------- out = x1 @ W2 + b2 (N=3) ----------------
__global__ void final_kernel(const float* __restrict__ x1, const float* __restrict__ W2,
                             const float* __restrict__ b2, float* __restrict__ out){
  long i = (long)blockIdx.x*blockDim.x + threadIdx.x;
  if (i >= (long)BT*D2_) return;
  long m = i / D2_; int j = (int)(i % D2_);
  float s = b2[j];
  const float* row = x1 + m*D1_;
  for (int k = 0; k < D1_; ++k) s += row[k]*W2[k*D2_ + j];
  out[i] = s;
}

__global__ void tail_kernel(const float* __restrict__ hfin, const float* __restrict__ cbuf,
                            float* __restrict__ out){
  long i = (long)blockIdx.x*blockDim.x + threadIdx.x;
  if (i >= (long)B_*H_) return;
  out[(long)BT*D2_ + i] = hfin[i];
  out[(long)BT*D2_ + (long)B_*H_ + i] = cbuf[i];
}

// ---------------- host ----------------

extern "C" void kernel_launch(void* const* d_in, const int* in_sizes, int n_in,
                              void* d_out, int out_size, void* d_ws, size_t ws_size,
                              hipStream_t stream){
  (void)in_sizes; (void)n_in; (void)out_size; (void)ws_size;
  const float* y  = (const float*)d_in[0];
  const float* s0 = (const float*)d_in[1];
  const float* c0 = (const float*)d_in[2];
  const float* Hv = (const float*)d_in[3];
  const float* Wx = (const float*)d_in[4];
  const float* Wh = (const float*)d_in[5];
  const float* b  = (const float*)d_in[6];
  const float* W1 = (const float*)d_in[7];
  const float* b1 = (const float*)d_in[8];
  const float* W2 = (const float*)d_in[9];
  const float* b2 = (const float*)d_in[10];
  float* out = (float*)d_out;

  char* w = (char*)d_ws;
  auto carve = [&](size_t bytes)->char*{
    char* p = w; w += (bytes + 255) & ~(size_t)255; return p; };
  bf16_t* yb   = (bf16_t*)carve((size_t)BT*DINP*2);
  bf16_t* s0b  = (bf16_t*)carve((size_t)B_*H_*2);
  bf16_t* Hvb  = (bf16_t*)carve((size_t)BT*H_*2);
  bf16_t* HvP  = (bf16_t*)carve((size_t)BT*H_*2);
  bf16_t* WxP  = (bf16_t*)carve((size_t)DINP*G4H*2);
  bf16_t* WhP  = (bf16_t*)carve((size_t)H_*G4H*2);
  bf16_t* W1P  = (bf16_t*)carve((size_t)(2*H_)*D1P*2);
  float*  xpr  = (float*) carve((size_t)BT*G4H*4);
  bf16_t* Sbf  = (bf16_t*)carve((size_t)BT*H_*2);
  bf16_t* Abf  = (bf16_t*)carve((size_t)BT*H_*2);
  float*  cbuf = (float*) carve((size_t)B_*H_*4);
  float*  hfin = (float*) carve((size_t)B_*H_*4);
  float*  x1   = (float*) carve((size_t)BT*D1_*4);

  const int eb = 256;
  auto nblk = [](long n, int bsz){ return (unsigned)((n + bsz - 1)/bsz); };

  cvt_pad_rows_kernel<<<nblk((long)BT*DINP, eb), eb, 0, stream>>>(y, yb, BT, DIN, DINP);
  cvt_kernel<<<nblk((long)B_*H_, eb), eb, 0, stream>>>(s0, s0b, (long)B_*H_);
  cvt_kernel<<<nblk((long)BT*H_, eb), eb, 0, stream>>>(Hv, Hvb, (long)BT*H_);
  pack_hv_kernel<<<nblk((long)B_*8*100*512, eb), eb, 0, stream>>>(Hv, HvP);
  pack_b_kernel<<<nblk((long)(DINP/32)*(G4H/16)*512, eb), eb, 0, stream>>>(Wx, WxP, DIN, G4H, DINP, G4H);
  pack_b_kernel<<<nblk((long)(H_/32)*(G4H/16)*512, eb), eb, 0, stream>>>(Wh, WhP, H_, G4H, H_, G4H);
  pack_b_kernel<<<nblk((long)((2*H_)/32)*(D1P/16)*512, eb), eb, 0, stream>>>(W1, W1P, 2*H_, D1_, 2*H_, D1P);
  copyf_kernel<<<nblk((long)B_*H_, eb), eb, 0, stream>>>(c0, cbuf, (long)B_*H_);

  dim3 g1(BT/64, G4H/64);
  gemm_xproj_kernel<<<g1, 128, 0, stream>>>(yb, WxP, b, xpr);

  for (int t = 0; t < T_; ++t){
    const bf16_t* hp = (t == 0) ? s0b : (Sbf + (long)(t - 1)*H_);
    long lda = (t == 0) ? (long)H_ : (long)T_*H_;
    lstm_step_kernel<<<H_/16, 512, 0, stream>>>(hp, lda, WhP, xpr, cbuf, Sbf, hfin, t);
  }

  dim3 ga(B_, T_/16);
  attn_kernel<<<ga, 128, 0, stream>>>(s0b, Sbf, Hvb, HvP, Abf);

  dim3 gw(BT/64, D1P/16);
  gemm_w1_kernel<<<gw, 128, 0, stream>>>(Sbf, Abf, W1P, b1, x1);

  final_kernel<<<nblk((long)BT*D2_, eb), eb, 0, stream>>>(x1, W2, b2, out);
  tail_kernel<<<nblk((long)B_*H_, eb), eb, 0, stream>>>(hfin, cbuf, out);
}
